// GTDampedOscillator_50148038148155
// MI455X (gfx1250) — matrix operand order repaired and checked by reading, not compiled, not run on hardware
//
#include <hip/hip_runtime.h>

typedef __attribute__((ext_vector_type(16))) __bf16 v16bf;
typedef __attribute__((ext_vector_type(8)))  float  v8f;

#define A_CH    4
#define M_OSC   32
#define S_LEN   48000
#define L_LEN   16000
#define CHUNK   4096
#define NCHUNK  12
#define WRPLEN  16640            // 240 zero-pad + 16000 taps + tail pad
#define SPPLEN  64512            // (L-1) lead pad + 48000 + tail pad
#define NTILE   188              // 188*256 = 48128 >= 48000
#define NSPLIT  8                // split-K over tap groups: 8 waves x 8 g-groups = 64
#define NG      (64 / NSPLIT)    // g-groups per wave
#define CPELEM  320              // elements per phase copy (272 used, padded for static loop)
#define TWO_PI  6.28318530717958647692f

__device__ __forceinline__ float softplusf(float x) {
    return fmaxf(x, 0.f) + log1pf(__expf(-fabsf(x)));   // stable softplus
}
__device__ __forceinline__ unsigned short f2bf(float f) {
    unsigned u = __float_as_uint(f);
    u += 0x7fffu + ((u >> 16) & 1u);                    // round-to-nearest-even
    return (unsigned short)(u >> 16);
}

// ---------------- K0a: per-mode parameters (tiny) ----------------
__global__ void k0a_params(const float* __restrict__ flin_p, const float* __restrict__ alpha_p,
                           const float* __restrict__ beta_p, const float* __restrict__ amp_v,
                           float* __restrict__ flinA, float* __restrict__ alA,
                           float* __restrict__ beA, float* __restrict__ ampA) {
    int tid = threadIdx.x;  // 128 threads
    if (tid < 32) {
        float ss = 0.f, ws = 0.f;
        #pragma unroll
        for (int j = 0; j < 8; ++j) {
            float Fv = 20.f * __expf(0.85592350f * (float)j);    // geomspace(20,8000,8)
            float s = softplusf(flin_p[tid * 8 + j]);
            ss += s; ws += s * Fv;
        }
        flinA[tid] = ws / ss;
        float ssa = 0.f, wsa = 0.f, ssb = 0.f, wsb = 0.f;
        for (int i = 0; i < 64; ++i) {
            float e  = __expf((float)i * (6.9077553f / 63.f));   // ln(1000)/63
            float sa = softplusf(alpha_p[tid * 64 + i]);
            float sb = softplusf(beta_p [tid * 64 + i]);
            ssa += sa; wsa += sa * (0.3f   * e);
            ssb += sb; wsb += sb * (1e-10f * e);
        }
        alA[tid] = wsa / ssa;
        beA[tid] = wsb / ssb;
    }
    if (tid < 128) {
        float s = 1.f / (1.f + __expf(-amp_v[tid]));
        ampA[tid] = 2.f * __powf(s, 2.30258509f) + 1e-7f;        // modified_sigmoid
    }
}

// ---------------- K0b: reversed padded bf16 weights + zero signal buffer ----------------
__global__ void k0b_pack(const float* __restrict__ forces,
                         unsigned short* __restrict__ wrp, unsigned short* __restrict__ spp) {
    int idx = blockIdx.x * 256 + threadIdx.x;
    const int WTOT = A_CH * WRPLEN;
    if (idx < WTOT) {
        int a = idx / WRPLEN, i = idx % WRPLEN;
        int j = i - 240;                       // tap index j'; wr[j'] = forces[L-1-j']
        float v = (j >= 0 && j < L_LEN) ? forces[a * L_LEN + (L_LEN - 1 - j)] : 0.f;
        wrp[idx] = f2bf(v);
    } else if (idx < WTOT + A_CH * SPPLEN) {
        spp[idx - WTOT] = 0;                   // zero pad; interior overwritten by K3
    }
}

// ---------------- K1: oscillator math + intra-chunk inclusive prefix ----------------
__global__ void k1_osc(const float* __restrict__ fnl, const float* __restrict__ flinA,
                       const float* __restrict__ alA, const float* __restrict__ beA,
                       float* __restrict__ dampPre, float* __restrict__ freqPre) {
    int blk = blockIdx.x;                      // (a*32+m)*NCHUNK + c
    int c   = blk % NCHUNK;
    int ch  = blk / NCHUNK;                    // 0..127 = a*32+m
    int m   = ch & 31;
    int tid = threadIdx.x;
    int t0  = c * CHUNK + tid * 16;

    float Fv[8];
    #pragma unroll
    for (int i = 0; i < 8; ++i) Fv[i] = 20.f * __expf(0.85592350f * (float)i);
    float fl = flinA[m], al = alA[m], be = beA[m];

    const float* base = fnl + (size_t)ch * S_LEN * 8;
    float dloc[16], floc[16];
    float dsum = 0.f, fsum = 0.f;
    #pragma unroll
    for (int i = 0; i < 16; ++i) {
        int t = t0 + i;
        float dv = 0.f, fv = 0.f;
        if (t < S_LEN) {
            const float4* p4 = (const float4*)(base + (size_t)t * 8);
            float4 x0 = p4[0], x1 = p4[1];
            float xs[8] = {x0.x, x0.y, x0.z, x0.w, x1.x, x1.y, x1.z, x1.w};
            float ss = 0.f, ws = 0.f;
            #pragma unroll
            for (int j = 0; j < 8; ++j) { float s = softplusf(xs[j]); ss += s; ws += s * Fv[j]; }
            float u   = fl + 0.3f * (ws / ss);
            float lbd = u * TWO_PI; lbd *= lbd;
            float dmp = 0.5f * (al + be * lbd);
            float fr  = sqrtf(fmaxf(lbd - dmp * dmp, 0.f)) * (1.f / TWO_PI);
            dv = dmp * (1.f / 48000.f);
            fv = fr  * (1.f / 48000.f);
        }
        dsum += dv; dloc[i] = dsum;
        fsum += fv; floc[i] = fsum;
    }
    __shared__ float sD[256], sF[256];
    sD[tid] = dsum; sF[tid] = fsum; __syncthreads();
    for (int off = 1; off < 256; off <<= 1) {   // Hillis-Steele block scan
        float vd = (tid >= off) ? sD[tid - off] : 0.f;
        float vf = (tid >= off) ? sF[tid - off] : 0.f;
        __syncthreads();
        sD[tid] += vd; sF[tid] += vf; __syncthreads();
    }
    float oD = sD[tid] - dsum, oF = sF[tid] - fsum;  // exclusive thread offsets
    size_t cb = (size_t)ch * S_LEN;
    #pragma unroll
    for (int i = 0; i < 16; ++i) {
        int t = t0 + i;
        if (t < S_LEN) { dampPre[cb + t] = dloc[i] + oD; freqPre[cb + t] = floc[i] + oF; }
    }
}

// ---------------- K2: chunk-offset scan over 128 chains ----------------
__global__ void k2_scan(const float* __restrict__ dampPre, const float* __restrict__ freqPre,
                        float* __restrict__ offD, float* __restrict__ offF) {
    int ch = threadIdx.x;                       // 128 chains
    size_t base = (size_t)ch * S_LEN;
    float d = 0.f, f = 0.f;
    for (int c = 0; c < NCHUNK; ++c) {
        offD[ch * NCHUNK + c] = d;
        offF[ch * NCHUNK + c] = f;
        int end = min(S_LEN, (c + 1) * CHUNK) - 1;
        d += dampPre[base + end];               // chunk total = last inclusive value
        f += freqPre[base + end];
    }
}

// ---------------- K3: combine scan, exp*sin, sum over modes, emit bf16 signal ----------------
__global__ void k3_signal(const float* __restrict__ dampPre, const float* __restrict__ freqPre,
                          const float* __restrict__ offD, const float* __restrict__ offF,
                          const float* __restrict__ ampA, unsigned short* __restrict__ spp) {
    int a = blockIdx.x / NTILE;
    int t = (blockIdx.x % NTILE) * 256 + threadIdx.x;
    if (t >= S_LEN) return;
    int ck = t >> 12;                           // CHUNK = 4096
    float acc = 0.f;
    for (int m = 0; m < M_OSC; ++m) {
        int ch = a * M_OSC + m;
        size_t idx = (size_t)ch * S_LEN + t;
        float cd = dampPre[idx] + offD[ch * NCHUNK + ck];
        float cf = freqPre[idx] + offF[ch * NCHUNK + ck];
        acc += ampA[ch] * __expf(-cd) * __sinf(TWO_PI * cf);
    }
    spp[(size_t)a * SPPLEN + (L_LEN - 1) + t] = f2bf(acc);
}

// ---------------- K4: skewed-Toeplitz WMMA FIR, split-K, 2-deep software pipeline ----------
// Wave owns one 256-output tile for one of NSPLIT disjoint tap ranges.
// C[m,n] -> out[T + 240 - 16m + n]; per g-group: 8 K=32 WMMA steps.
// A = 16-row-shifted reversed weights (aligned b128 global loads, L2-resident).
// B = Toeplitz signal via 4 phase-shifted per-wave LDS copies (aligned ds_load_b64),
//     ping-pong staged: global->regs latency hidden by burst N, regs->LDS in burst N+1.
__global__ void k4_fir(const unsigned short* __restrict__ wrp,
                       const unsigned short* __restrict__ spp,
                       float* __restrict__ partial) {
    __shared__ __align__(16) unsigned short lds[8 * 2 * 4 * CPELEM];  // per-wave ping-pong
    int wave = threadIdx.x >> 5;
    int lane = threadIdx.x & 31;
    int gw   = blockIdx.x * 8 + wave;            // 0 .. NSPLIT*752-1
    int s    = gw / (A_CH * NTILE);
    int rem  = gw % (A_CH * NTILE);
    int a    = rem / NTILE;
    int T    = (rem % NTILE) * 256;
    int g0   = s * NG;

    const unsigned short* sp = spp + (size_t)a * SPPLEN;
    const int4* wrq = (const int4*)(wrp + (size_t)a * WRPLEN);        // 16B-aligned
    unsigned short* base = lds + wave * (2 * 4 * CPELEM);
    unsigned short* bufs[2] = { base, base + 4 * CPELEM };

    int lm = lane & 15, hi = lane >> 4;
    v8f C = {};
    union AF { v16bf v; int4 q[2]; } Af[2];
    union BF { v16bf v; uint2 q[4]; } Bf[2];
    unsigned st[20];                             // staged window (4 phases x 5 slots)

    auto loadA = [&](int step, int k) {
        int qi = 4 * step + 2 * lm + 2 * hi;
        Af[k].q[0] = wrq[qi]; Af[k].q[1] = wrq[qi + 1];
    };
    auto loadB = [&](const unsigned short* cb, int c2, int k) {
        int r0 = 32 * c2 + lm + 16 * hi, ph = r0 & 3;
        const uint2* bs = (const uint2*)(cb + ph * CPELEM + (r0 - ph));   // 8B-aligned
        Bf[k].q[0] = bs[0]; Bf[k].q[1] = bs[1]; Bf[k].q[2] = bs[2]; Bf[k].q[3] = bs[3];
    };
    auto stageLoad = [&](int g) {                // global window -> registers
        int wb = T + 256 * g;
        #pragma unroll
        for (int phi = 0; phi < 4; ++phi)
            #pragma unroll
            for (int j = 0; j < 5; ++j) {
                const unsigned short* q = sp + wb + phi + j * 64 + 2 * lane;
                st[phi * 5 + j] = (unsigned)q[0] | ((unsigned)q[1] << 16);
            }
    };
    auto stageStore = [&](unsigned short* dst) { // registers -> LDS phase copies
        #pragma unroll
        for (int phi = 0; phi < 4; ++phi)
            #pragma unroll
            for (int j = 0; j < 5; ++j)
                *(unsigned*)(dst + phi * CPELEM + j * 64 + 2 * lane) = st[phi * 5 + j];
    };

    // prologue: cold-stage group g0; put g0+1's window in flight; prime fragment 0
    stageLoad(g0); stageStore(bufs[0]);
    stageLoad(g0 + 1);
    loadA(8 * g0, 0); loadB(bufs[0], 0, 0);

    #pragma unroll
    for (int gg = 0; gg < NG; ++gg) {
        const unsigned short* cb = bufs[gg & 1];
        int g = g0 + gg;
        #pragma unroll
        for (int c2 = 0; c2 < 8; ++c2) {         // burst: 8 WMMAs, fragments double-buffered
            int cur = c2 & 1, nxt = cur ^ 1;
            if (c2 < 7) { loadA(8 * g + c2 + 1, nxt); loadB(cb, c2 + 1, nxt); }
            C = __builtin_amdgcn_wmma_f32_16x16x32_bf16(false, Af[cur].v, false, Bf[cur].v,
                                                        (short)0, C, false, false);
        }
        if (gg < NG - 1) {
            unsigned short* nb = bufs[(gg + 1) & 1];
            stageStore(nb);                      // window g+1 (loads covered by this burst)
            if (gg < NG - 2) stageLoad(g + 2);   // next window: covered by next burst
            loadA(8 * (g + 1), 0); loadB(nb, 0, 0);   // prime fragment 0 of next burst
        }
    }
    #pragma unroll
    for (int r = 0; r < 8; ++r) {
        int m = r + 8 * hi;                      // D elem r -> row M = r + 8*hi, col N = lm
        int t = T + 240 - 16 * m + lm;
        if (t < S_LEN) partial[((size_t)s * A_CH + a) * S_LEN + t] = C[r];
    }
}

// ---------------- K5: deterministic fixed-order reduction of split-K partials ----------------
__global__ void k5_reduce(const float* __restrict__ partial, float* __restrict__ out) {
    int idx = blockIdx.x * 256 + threadIdx.x;    // a*S_LEN + t
    if (idx >= A_CH * S_LEN) return;
    float acc = 0.f;
    #pragma unroll
    for (int s = 0; s < NSPLIT; ++s) acc += partial[(size_t)s * A_CH * S_LEN + idx];
    out[idx] = acc;
}

extern "C" void kernel_launch(void* const* d_in, const int* in_sizes, int n_in,
                              void* d_out, int out_size, void* d_ws, size_t ws_size,
                              hipStream_t stream) {
    const float* flin_p  = (const float*)d_in[0];   // (1,32,1,8)
    const float* fnl_p   = (const float*)d_in[1];   // (4,32,48000,8)
    const float* alpha_p = (const float*)d_in[2];   // (1,32,1,64)
    const float* beta_p  = (const float*)d_in[3];   // (1,32,1,64)
    const float* amp_v   = (const float*)d_in[4];   // (4,32,1)
    const float* forces  = (const float*)d_in[5];   // (4,16000)
    float* out = (float*)d_out;                     // (4,48000) fp32

    char* w = (char*)d_ws;
    float* dampPre = (float*)w; w += (size_t)A_CH * M_OSC * S_LEN * 4;
    float* freqPre = (float*)w; w += (size_t)A_CH * M_OSC * S_LEN * 4;
    float* offD    = (float*)w; w += 128 * NCHUNK * 4;
    float* offF    = (float*)w; w += 128 * NCHUNK * 4;
    float* ampA    = (float*)w; w += 128 * 4;
    float* flinA   = (float*)w; w += 32 * 4;
    float* alA     = (float*)w; w += 32 * 4;
    float* beA     = (float*)w; w += 32 * 4;
    unsigned short* wrp = (unsigned short*)w; w += (size_t)A_CH * WRPLEN * 2;
    unsigned short* spp = (unsigned short*)w; w += (size_t)A_CH * SPPLEN * 2;
    float* partial = (float*)w;                      // NSPLIT * A_CH * S_LEN floats

    k0a_params<<<1, 128, 0, stream>>>(flin_p, alpha_p, beta_p, amp_v, flinA, alA, beA, ampA);

    int n0b = A_CH * WRPLEN + A_CH * SPPLEN;
    k0b_pack<<<(n0b + 255) / 256, 256, 0, stream>>>(forces, wrp, spp);

    k1_osc<<<A_CH * M_OSC * NCHUNK, 256, 0, stream>>>(fnl_p, flinA, alA, beA, dampPre, freqPre);

    k2_scan<<<1, 128, 0, stream>>>(dampPre, freqPre, offD, offF);

    k3_signal<<<A_CH * NTILE, 256, 0, stream>>>(dampPre, freqPre, offD, offF, ampA, spp);

    k4_fir<<<(NSPLIT * A_CH * NTILE) / 8, 256, 0, stream>>>(wrp, spp, partial);

    k5_reduce<<<(A_CH * S_LEN + 255) / 256, 256, 0, stream>>>(partial, out);
}
